// Model_39676907882004
// MI455X (gfx1250) — matrix operand order repaired and checked by reading, not compiled, not run on hardware
//
#include <hip/hip_runtime.h>
#include <hip/hip_bf16.h>
#include <math.h>

typedef __attribute__((ext_vector_type(16))) __bf16 v16bf;
typedef __attribute__((ext_vector_type(8)))  float  v8f;

#define S_DIM  4096
#define D_DIM  1024
#define DV_DIM 1024

// ---------------- conversion / scale prep ----------------

__global__ void cvt_f32_to_bf16(const float* __restrict__ in,
                                __bf16* __restrict__ out, int n4) {
    int i = blockIdx.x * blockDim.x + threadIdx.x;   // one float4 per thread
    if (i >= n4) return;
    float4 f = ((const float4*)in)[i];
    union { unsigned int u[2]; __bf16 h[4]; } cv;
    cv.h[0] = (__bf16)f.x; cv.h[1] = (__bf16)f.y;
    cv.h[2] = (__bf16)f.z; cv.h[3] = (__bf16)f.w;
    ((uint2*)out)[i] = make_uint2(cv.u[0], cv.u[1]);
}

__global__ void build_scale(const float* __restrict__ q,
                            float* __restrict__ scale) {
    int j = blockIdx.x * blockDim.x + threadIdx.x;   // 0..S-1
    if (j >= S_DIM) return;
    int group = j / D_DIM;               // 0..3
    int col   = j % D_DIM;
    // q[:, ::step] rows: group*1024 ; element offset row*D + col
    scale[j] = q[(size_t)group * (S_DIM / 4) * D_DIM + col];
}

// ---------------- WMMA fragment helpers ----------------

union FragU { v16bf v; unsigned int u[8]; };

// A-matrix 16x32 bf16 layout: lane = M(0..15) in each half-wave;
// lanes 0-15 hold K {0..7,16..23}, lanes 16-31 hold K {8..15,24..31},
// VGPR j holds a K-pair (2 consecutive K).
__device__ __forceinline__ void load_a_frag(FragU& a, const __bf16* row, int khalf) {
#pragma unroll
    for (int j = 0; j < 8; ++j) {
        int kk = (j < 4) ? (khalf * 8 + 2 * j) : (16 + khalf * 8 + 2 * (j - 4));
        a.u[j] = *(const unsigned int*)(row + kk);
    }
}

// B-matrix 32x16 bf16 layout: lane = N(0..15) in each half-wave;
// lanes 0-15 hold K {0..7,16..23}, lanes 16-31 hold K {8..15,24..31}; VGPR j holds a K-pair.
__device__ __forceinline__ void load_b_frag(FragU& b, const __bf16* row, int khalf) {
#pragma unroll
    for (int j = 0; j < 8; ++j) {
        int kk = (j < 4) ? (khalf * 8 + 2 * j) : (16 + khalf * 8 + 2 * (j - 4));
        b.u[j] = *(const unsigned int*)(row + kk);
    }
}

// ---------------- GEMM tiling ----------------
// block tile 128(M) x 128(N), 8 waves arranged 4(M) x 2(N),
// wave tile 32(M) x 64(N)  -> 8 WMMA per wave per K-step (B frags reused 2x)

#define BM 128
#define BN 128
#define BK 32
#define LDT 34   // padded LDS row stride (elements); 17 dwords -> conflict-free

// ---------------- kernel 1: logits = (Q K^T) * scale[col], bf16 out ----------------

__global__ __launch_bounds__(256)
void gemm_qk(const __bf16* __restrict__ qb, const __bf16* __restrict__ kb,
             const float* __restrict__ scale, __bf16* __restrict__ logits) {
    __shared__ __bf16 Qs[BM * LDT];
    __shared__ __bf16 Ks[BN * LDT];

    const int tid  = threadIdx.x;
    const int lane = tid & 31;
    const int wave = tid >> 5;
    const int wm   = wave & 3;    // M sub-tile: wm*32
    const int wn   = wave >> 2;   // N sub-tile: wn*64
    const int m0 = blockIdx.y * BM;
    const int n0 = blockIdx.x * BN;

    v8f acc[2][4];
#pragma unroll
    for (int r = 0; r < 2; ++r)
#pragma unroll
        for (int g = 0; g < 4; ++g)
#pragma unroll
            for (int j = 0; j < 8; ++j) acc[r][g][j] = 0.f;

    const int nlane = lane & 15;
    const int khalf = lane >> 4;

    for (int k0 = 0; k0 < D_DIM; k0 += BK) {
        // stage Q tile: 128x32 bf16, 8-byte chunks, fully coalesced
#pragma unroll
        for (int i = 0; i < 4; ++i) {
            int chunk = tid + i * 256;
            int r = chunk >> 3;
            int c = (chunk & 7) * 4;
            uint2 val = *(const uint2*)(qb + (size_t)(m0 + r) * D_DIM + k0 + c);
            unsigned int* dst = (unsigned int*)(Qs + r * LDT + c);
            dst[0] = val.x; dst[1] = val.y;
        }
        // stage K tile: 128x32 bf16 (rows = output columns)
#pragma unroll
        for (int i = 0; i < 4; ++i) {
            int chunk = tid + i * 256;
            int r = chunk >> 3;
            int c = (chunk & 7) * 4;
            uint2 val = *(const uint2*)(kb + (size_t)(n0 + r) * D_DIM + k0 + c);
            unsigned int* dst = (unsigned int*)(Ks + r * LDT + c);
            dst[0] = val.x; dst[1] = val.y;
        }
        __syncthreads();

        FragU a0, a1;
        load_a_frag(a0, Qs + (wm * 32 +      nlane) * LDT, khalf);
        load_a_frag(a1, Qs + (wm * 32 + 16 + nlane) * LDT, khalf);
#pragma unroll
        for (int g = 0; g < 4; ++g) {
            FragU b;
            load_b_frag(b, Ks + (wn * 64 + g * 16 + nlane) * LDT, khalf);
            acc[0][g] = __builtin_amdgcn_wmma_f32_16x16x32_bf16(
                false, a0.v, false, b.v, (short)0, acc[0][g], false, false);
            acc[1][g] = __builtin_amdgcn_wmma_f32_16x16x32_bf16(
                false, a1.v, false, b.v, (short)0, acc[1][g], false, false);
        }
        __syncthreads();
    }

    // epilogue: multiply per-column scale, store bf16 logits
    const int mhalf = lane >> 4;
#pragma unroll
    for (int r = 0; r < 2; ++r)
#pragma unroll
        for (int g = 0; g < 4; ++g) {
            int nglob = n0 + wn * 64 + g * 16 + nlane;
            float sc = scale[nglob];
#pragma unroll
            for (int j = 0; j < 8; ++j) {
                int mglob = m0 + wm * 32 + r * 16 + mhalf * 8 + j;
                logits[(size_t)mglob * S_DIM + nglob] = (__bf16)(acc[r][g][j] * sc);
            }
        }
}

// ---------------- kernel 2: in-place row softmax over bf16 logits ----------------

__global__ __launch_bounds__(256)
void softmax_row(__bf16* __restrict__ logits) {
    const int row = blockIdx.x;
    const int tid = threadIdx.x;
    __bf16* p = logits + (size_t)row * S_DIM;

    float x[16];
    float m = -INFINITY;
#pragma unroll
    for (int i = 0; i < 16; ++i) {
        x[i] = (float)p[tid + i * 256];
        m = fmaxf(m, x[i]);
    }
    __shared__ float red[256];
    red[tid] = m;
    __syncthreads();
    for (int s = 128; s > 0; s >>= 1) {
        if (tid < s) red[tid] = fmaxf(red[tid], red[tid + s]);
        __syncthreads();
    }
    m = red[0];
    __syncthreads();

    float sum = 0.f;
#pragma unroll
    for (int i = 0; i < 16; ++i) {
        x[i] = __expf(x[i] - m);
        sum += x[i];
    }
    red[tid] = sum;
    __syncthreads();
    for (int s = 128; s > 0; s >>= 1) {
        if (tid < s) red[tid] += red[tid + s];
        __syncthreads();
    }
    float inv = 1.0f / red[0];
#pragma unroll
    for (int i = 0; i < 16; ++i)
        p[tid + i * 256] = (__bf16)(x[i] * inv);
}

// ---------------- kernel 3: out = P @ V, fp32 out ----------------

__global__ __launch_bounds__(256)
void gemm_pv(const __bf16* __restrict__ Pb, const __bf16* __restrict__ vb,
             float* __restrict__ out) {
    __shared__ __bf16 Ps[BM * LDT];
    __shared__ __bf16 Vs[BN * LDT];   // transposed: Vs[n_local][k_local]

    const int tid  = threadIdx.x;
    const int lane = tid & 31;
    const int wave = tid >> 5;
    const int wm   = wave & 3;
    const int wn   = wave >> 2;
    const int m0 = blockIdx.y * BM;
    const int n0 = blockIdx.x * BN;

    v8f acc[2][4];
#pragma unroll
    for (int r = 0; r < 2; ++r)
#pragma unroll
        for (int g = 0; g < 4; ++g)
#pragma unroll
            for (int j = 0; j < 8; ++j) acc[r][g][j] = 0.f;

    const int nlane = lane & 15;
    const int khalf = lane >> 4;

    for (int k0 = 0; k0 < S_DIM; k0 += BK) {
        // stage P tile: 128x32 bf16
#pragma unroll
        for (int i = 0; i < 4; ++i) {
            int chunk = tid + i * 256;
            int r = chunk >> 3;
            int c = (chunk & 7) * 4;
            uint2 val = *(const uint2*)(Pb + (size_t)(m0 + r) * S_DIM + k0 + c);
            unsigned int* dst = (unsigned int*)(Ps + r * LDT + c);
            dst[0] = val.x; dst[1] = val.y;
        }
        // stage V tile 32(K) x 128(N), written transposed so K-pairs are contiguous
#pragma unroll
        for (int i = 0; i < 4; ++i) {
            int chunk = tid + i * 256;
            int r = chunk >> 5;          // K index 0..31
            int c = (chunk & 31) * 4;    // N offset 0..124
            uint2 raw = *(const uint2*)(vb + (size_t)(k0 + r) * DV_DIM + n0 + c);
            union { uint2 u2; __bf16 h[4]; } cv; cv.u2 = raw;
#pragma unroll
            for (int t = 0; t < 4; ++t)
                Vs[(c + t) * LDT + r] = cv.h[t];
        }
        __syncthreads();

        FragU a0, a1;
        load_a_frag(a0, Ps + (wm * 32 +      nlane) * LDT, khalf);
        load_a_frag(a1, Ps + (wm * 32 + 16 + nlane) * LDT, khalf);
#pragma unroll
        for (int g = 0; g < 4; ++g) {
            FragU b;
            load_b_frag(b, Vs + (wn * 64 + g * 16 + nlane) * LDT, khalf);
            acc[0][g] = __builtin_amdgcn_wmma_f32_16x16x32_bf16(
                false, a0.v, false, b.v, (short)0, acc[0][g], false, false);
            acc[1][g] = __builtin_amdgcn_wmma_f32_16x16x32_bf16(
                false, a1.v, false, b.v, (short)0, acc[1][g], false, false);
        }
        __syncthreads();
    }

    const int mhalf = lane >> 4;
#pragma unroll
    for (int r = 0; r < 2; ++r)
#pragma unroll
        for (int g = 0; g < 4; ++g) {
            int nglob = n0 + wn * 64 + g * 16 + nlane;
#pragma unroll
            for (int j = 0; j < 8; ++j) {
                int mglob = m0 + wm * 32 + r * 16 + mhalf * 8 + j;
                out[(size_t)mglob * DV_DIM + nglob] = acc[r][g][j];
            }
        }
}

// ---------------- host launch ----------------

extern "C" void kernel_launch(void* const* d_in, const int* in_sizes, int n_in,
                              void* d_out, int out_size, void* d_ws, size_t ws_size,
                              hipStream_t stream) {
    const float* q = (const float*)d_in[0];
    const float* k = (const float*)d_in[1];
    const float* v = (const float*)d_in[2];
    // d_in[3] scale_factor, d_in[4] dropout_p: overwritten/ignored by reference
    float* out = (float*)d_out;

    // workspace layout (~58 MB)
    __bf16* qb    = (__bf16*)d_ws;                       // S*D
    __bf16* kb    = qb + (size_t)S_DIM * D_DIM;          // S*D
    __bf16* vb    = kb + (size_t)S_DIM * D_DIM;          // S*DV
    float*  scale = (float*)(vb + (size_t)S_DIM * DV_DIM);   // S floats (4B aligned)
    __bf16* Pb    = (__bf16*)(scale + S_DIM);            // S*S bf16 logits -> probs

    const int threads = 256;

    int n4 = (S_DIM * D_DIM) / 4;
    cvt_f32_to_bf16<<<(n4 + threads - 1) / threads, threads, 0, stream>>>(q, qb, n4);
    cvt_f32_to_bf16<<<(n4 + threads - 1) / threads, threads, 0, stream>>>(k, kb, n4);
    int nv4 = (S_DIM * DV_DIM) / 4;
    cvt_f32_to_bf16<<<(nv4 + threads - 1) / threads, threads, 0, stream>>>(v, vb, nv4);

    build_scale<<<(S_DIM + threads - 1) / threads, threads, 0, stream>>>(q, scale);

    dim3 gq(S_DIM / BN, S_DIM / BM);     // (32, 32)
    gemm_qk<<<gq, threads, 0, stream>>>(qb, kb, scale, Pb);

    softmax_row<<<S_DIM, threads, 0, stream>>>(Pb);

    dim3 gp(DV_DIM / BN, S_DIM / BM);    // (8, 32)
    gemm_pv<<<gp, threads, 0, stream>>>(Pb, vb, out);
}
